// GATv2_GCN_44925357916176
// MI455X (gfx1250) — matrix operand order repaired and checked by reading, not compiled, not run on hardware
//
#include <hip/hip_runtime.h>
#include <hip/hip_bf16.h>
#include <stdint.h>

// ---------------------------------------------------------------------------
// GATv2 + GCN + pool + 2xFC for MI455X (gfx1250, wave32, WMMA)
// ---------------------------------------------------------------------------

typedef _Float16 half_t;
typedef __attribute__((ext_vector_type(16))) _Float16 v16h;
typedef __attribute__((ext_vector_type(8)))  _Float16 v8h;
typedef __attribute__((ext_vector_type(8)))  float    v8f;

#define HEADS 10
#define CH    78
#define F     780      // HEADS*CH
#define GRAPHS 512
#define POOLF  1560
#define FC1N   1500
#define FC2N   128

// ----- ordered-uint encoding for float atomic max -----
__device__ __forceinline__ unsigned ordF(float f) {
  unsigned u = __float_as_uint(f);
  return (u & 0x80000000u) ? ~u : (u | 0x80000000u);
}
__device__ __forceinline__ float unordF(unsigned u) {
  return (u & 0x80000000u) ? __uint_as_float(u ^ 0x80000000u)
                           : __uint_as_float(~u);
}

// ---------------------------------------------------------------------------
// A converter: f32 [M x K] -> zero-padded f16 [M x Kp]
// ---------------------------------------------------------------------------
__global__ void convA(const float* __restrict__ A, half_t* __restrict__ A16,
                      long long M, int K, int Kp)
{
  long long idx = (long long)blockIdx.x * blockDim.x + threadIdx.x;
  if (idx >= M * Kp) return;
  long long m = idx / Kp;
  int k = (int)(idx - m * Kp);
  A16[idx] = (half_t)((k < K) ? A[m * K + k] : 0.f);
}

// ---------------------------------------------------------------------------
// B packer: f32 [K x N] -> fragment-order f16.
// Layout: block (nt, kt) of 512 halfs at offset ((nt*KT + kt)*32 + lane)*16.
// lane l, half i  <-  B[kt*32 + (i&7) + 16*(i>=8) + 8*(l>=16)][nt*16 + (l&15)]  (zero pad)
// A wave's B fragment for subtile nt, step kt is then 32 contiguous bytes/lane.
// ---------------------------------------------------------------------------
__global__ void packB(const float* __restrict__ B, half_t* __restrict__ P,
                      int K, int N, int Kp, int Np)
{
  long long idx = (long long)blockIdx.x * blockDim.x + threadIdx.x;
  long long tot = (long long)Np * Kp;
  if (idx >= tot) return;
  int i    = (int)(idx & 15);
  int lane = (int)((idx >> 4) & 31);
  long long t = idx >> 9;                 // nt*KT + kt
  int KT = Kp >> 5;
  int nt = (int)(t / KT);
  int kt = (int)(t - (long long)nt * KT);
  int k = kt * 32 + (i & 7) + ((i >> 3) << 4) + ((lane >> 4) << 3);
  int n = nt * 16 + (lane & 15);
  float v = (k < K && n < N) ? B[(size_t)k * N + n] : 0.f;
  P[idx] = (half_t)v;
}

// ---------------------------------------------------------------------------
// WMMA GEMM:  C[M,N] = act(A16[M,Kp] @ B + bias)
// Block = 256 threads = 8 waves; wave tile 16x64 (4 N-subtiles), block 128x64.
// No LDS, no branches in the K loop: A chunks and packed-B fragments are
// contiguous global b128 loads; weights/activations live in L2 (192 MB).
// ---------------------------------------------------------------------------
__global__ __launch_bounds__(256)
void wmma_gemm(const half_t* __restrict__ A16, const half_t* __restrict__ Bpk,
               const float* __restrict__ bias, float* __restrict__ C,
               int M, int N, int Kp, int relu)
{
  const int tid  = threadIdx.x;
  const int wave = tid >> 5;
  const int lane = tid & 31;
  const int hi   = lane >> 4;       // lane half
  const int lr   = lane & 15;

  const int rowA = blockIdx.x * 128 + wave * 16 + lr;   // M multiple of 128
  const int nt0  = blockIdx.y * 4;                      // 4 subtiles of 16 cols
  const int KT   = Kp >> 5;

  const half_t* Ap = A16 + (size_t)rowA * Kp + hi * 8;
  const half_t* Bp = Bpk + ((size_t)nt0 * KT) * 512 + (size_t)lane * 16;
  const size_t  bStride = (size_t)KT * 512;             // per N-subtile

  v8f acc[4] = {};

  for (int kt = 0; kt < KT; ++kt) {
    // A fragment: two contiguous 16B chunks per lane (16-bit A 16x32 layout)
    v8h a0 = *(const v8h*)(Ap + kt * 32);        // K = hi*8 .. +7
    v8h a1 = *(const v8h*)(Ap + kt * 32 + 16);   // K = 16+hi*8 .. +7
    v16h a = __builtin_shufflevector(a0, a1, 0, 1, 2, 3, 4, 5, 6, 7,
                                             8, 9, 10, 11, 12, 13, 14, 15);
    const half_t* bb = Bp + (size_t)kt * 512;
    #pragma unroll
    for (int s = 0; s < 4; ++s) {
      v16h b = *(const v16h*)(bb + s * bStride);
      acc[s] = __builtin_amdgcn_wmma_f32_16x16x32_f16(
          false, a, false, b, (short)0, acc[s], false, false);
    }
  }

  // D layout: VGPR i -> row M = i + 8*hi, col = lane&15
  #pragma unroll
  for (int s = 0; s < 4; ++s) {
    int col = (nt0 + s) * 16 + lr;
    if (col < N) {
      float bv = bias ? bias[col] : 0.f;
      #pragma unroll
      for (int i = 0; i < 8; ++i) {
        int row = blockIdx.x * 128 + wave * 16 + i + hi * 8;
        float v = acc[s][i] + bv;
        if (relu) v = v > 0.f ? v : 0.f;
        C[(size_t)row * N + col] = v;
      }
    }
  }
}

// ---------------------------------------------------------------------------
// Utility fills
// ---------------------------------------------------------------------------
__global__ void fill_f32(float* p, float v, long long n) {
  long long i = (long long)blockIdx.x * blockDim.x + threadIdx.x;
  if (i < n) p[i] = v;
}

// ---------------------------------------------------------------------------
// GATv2 attention: logits + per-dst running max (ordered-uint atomic max)
// ---------------------------------------------------------------------------
__global__ __launch_bounds__(256)
void attn_logits(const float* __restrict__ XL, const float* __restrict__ XR,
                 const int* __restrict__ ei, const float* __restrict__ att,
                 float* __restrict__ P, unsigned* __restrict__ mOrd,
                 int E, int N)
{
  int e = blockIdx.x * 8 + (threadIdx.x >> 5);
  int lane = threadIdx.x & 31;
  int Etot = E + N;
  if (e >= Etot) return;
  int src = (e < E) ? ei[e]     : (e - E);
  int dst = (e < E) ? ei[E + e] : (e - E);
  const float* xl = XL + (size_t)src * F;
  const float* xr = XR + (size_t)dst * F;
  #pragma unroll 1
  for (int h = 0; h < HEADS; ++h) {
    float s = 0.f;
    for (int c = lane; c < CH; c += 32) {
      float v = xl[h * CH + c] + xr[h * CH + c];
      v = (v > 0.f) ? v : 0.2f * v;           // LeakyReLU(0.2)
      s += att[h * CH + c] * v;
    }
    #pragma unroll
    for (int off = 16; off > 0; off >>= 1) s += __shfl_xor(s, off, 32);
    if (lane == 0) {
      P[(size_t)e * HEADS + h] = s;
      atomicMax(&mOrd[dst * HEADS + h], ordF(s));
    }
  }
}

__global__ void attn_exp(const int* __restrict__ ei,
                         const unsigned* __restrict__ mOrd,
                         float* __restrict__ P, float* __restrict__ denom,
                         int E, int N)
{
  long long idx = (long long)blockIdx.x * blockDim.x + threadIdx.x;
  long long tot = (long long)(E + N) * HEADS;
  if (idx >= tot) return;
  int e = (int)(idx / HEADS);
  int h = (int)(idx - (long long)e * HEADS);
  int dst = (e < E) ? ei[E + e] : (e - E);
  float m = unordF(mOrd[dst * HEADS + h]);
  float p = __expf(P[idx] - m);
  P[idx] = p;
  atomicAdd(&denom[dst * HEADS + h], p);
}

__global__ __launch_bounds__(256)
void attn_agg(const float* __restrict__ XL, const int* __restrict__ ei,
              const float* __restrict__ P, const float* __restrict__ denom,
              float* __restrict__ H1, int E, int N)
{
  int e = blockIdx.x;
  int src = (e < E) ? ei[e]     : (e - E);
  int dst = (e < E) ? ei[E + e] : (e - E);
  __shared__ float alpha[HEADS];
  if (threadIdx.x < HEADS)
    alpha[threadIdx.x] = P[(size_t)e * HEADS + threadIdx.x] /
                         denom[dst * HEADS + threadIdx.x];
  __syncthreads();
  const float* xs = XL + (size_t)src * F;
  float* hd = H1 + (size_t)dst * F;
  for (int j = threadIdx.x; j < F; j += blockDim.x)
    atomicAdd(&hd[j], alpha[j / CH] * xs[j]);
}

__global__ void bias_relu(float* __restrict__ X, const float* __restrict__ b,
                          long long rows, int cols, int relu)
{
  long long idx = (long long)blockIdx.x * blockDim.x + threadIdx.x;
  if (idx >= rows * cols) return;
  int j = (int)(idx % cols);
  float v = X[idx] + b[j];
  if (relu) v = v > 0.f ? v : 0.f;
  X[idx] = v;
}

__global__ void deg_kernel(const int* __restrict__ ei, float* __restrict__ deg,
                           int E, int N)
{
  int e = blockIdx.x * blockDim.x + threadIdx.x;
  if (e >= E + N) return;
  int dst = (e < E) ? ei[E + e] : (e - E);
  atomicAdd(&deg[dst], 1.f);
}

__global__ __launch_bounds__(256)
void gcn_agg(const float* __restrict__ Hg, const float* __restrict__ deg,
             const int* __restrict__ ei, float* __restrict__ H2, int E, int N)
{
  int e = blockIdx.x;
  int src = (e < E) ? ei[e]     : (e - E);
  int dst = (e < E) ? ei[E + e] : (e - E);
  float ds = deg[src], dd = deg[dst];
  float norm = ((ds > 0.f) ? rsqrtf(ds) : 0.f) *
               ((dd > 0.f) ? rsqrtf(dd) : 0.f);
  const float* hs = Hg + (size_t)src * F;
  float* hd = H2 + (size_t)dst * F;
  for (int j = threadIdx.x; j < F; j += blockDim.x)
    atomicAdd(&hd[j], norm * hs[j]);
}

__global__ void pool_cnt(const int* __restrict__ batch, float* __restrict__ cnt,
                         int N)
{
  int i = blockIdx.x * blockDim.x + threadIdx.x;
  if (i < N) atomicAdd(&cnt[batch[i]], 1.f);
}

__global__ void pool_accum(const float* __restrict__ H2,
                           const int* __restrict__ batch,
                           unsigned* __restrict__ pmax,
                           float* __restrict__ psum, int N)
{
  long long idx = (long long)blockIdx.x * blockDim.x + threadIdx.x;
  if (idx >= (long long)N * F) return;
  int node = (int)(idx / F);
  int j = (int)(idx - (long long)node * F);
  int g = batch[node];
  float v = H2[idx];
  atomicMax(&pmax[(size_t)g * F + j], ordF(v));
  atomicAdd(&psum[(size_t)g * F + j], v);
}

__global__ void pool_finalize(const unsigned* __restrict__ pmax,
                              const float* __restrict__ psum,
                              const float* __restrict__ cnt,
                              float* __restrict__ G)
{
  long long idx = (long long)blockIdx.x * blockDim.x + threadIdx.x;
  if (idx >= (long long)GRAPHS * F) return;
  int g = (int)(idx / F);
  int j = (int)(idx - (long long)g * F);
  float mx = unordF(pmax[idx]);
  if (!(mx > -__builtin_inff() && mx < __builtin_inff())) mx = 0.f;
  G[(size_t)g * POOLF + j]     = mx;
  G[(size_t)g * POOLF + F + j] = psum[idx] / fmaxf(cnt[g], 1.f);
}

// ---------------------------------------------------------------------------
// host-side orchestration
// ---------------------------------------------------------------------------
static inline unsigned blks(long long n, int t) {
  return (unsigned)((n + t - 1) / t);
}
static inline dim3 gemm_grid(int M, int Np64) {   // Np64 = N padded to 64
  return dim3((unsigned)(M / 128), (unsigned)(Np64 / 64));
}

extern "C" void kernel_launch(void* const* d_in, const int* in_sizes, int n_in,
                              void* d_out, int out_size, void* d_ws, size_t ws_size,
                              hipStream_t stream)
{
  const float* x    = (const float*)d_in[0];
  const int*   ei   = (const int*)d_in[1];
  const int*   batch= (const int*)d_in[2];
  const float* Wl   = (const float*)d_in[3];
  const float* Wr   = (const float*)d_in[4];
  const float* att  = (const float*)d_in[5];
  const float* bgat = (const float*)d_in[6];
  const float* Wg   = (const float*)d_in[7];
  const float* bg   = (const float*)d_in[8];
  const float* Wf1  = (const float*)d_in[9];
  const float* bf1  = (const float*)d_in[10];
  const float* Wf2  = (const float*)d_in[11];
  const float* bf2  = (const float*)d_in[12];
  float* out = (float*)d_out;

  const int N    = in_sizes[0] / CH;   // 16384
  const int E    = in_sizes[1] / 2;    // 262144
  const int Etot = E + N;

  // padded dims: Kp -> mult of 32, Np -> mult of 64
  const int Kp1 = 96,   Np1 = 832;    // GEMM1: K=78,   N=780
  const int KpG = 800,  NpG = 832;    // GCN:   K=780,  N=780
  const int Kf1 = 1568, Nf1 = 1536;   // FC1:   K=1560, N=1500
  const int Kf2 = 1504, Nf2 = 128;    // FC2:   K=1500, N=128

  // bump allocator over workspace
  char* ws = (char*)d_ws;
  size_t off = 0;
  auto alloc = [&](size_t bytes) -> char* {
    char* p = ws + off;
    off = (off + bytes + 255) & ~(size_t)255;
    return p;
  };
  float*    XL    = (float*)   alloc((size_t)N * F * 4);
  float*    XR    = (float*)   alloc((size_t)N * F * 4);  // reused as Hg
  float*    H1    = (float*)   alloc((size_t)N * F * 4);
  float*    P     = (float*)   alloc((size_t)Etot * HEADS * 4);
  unsigned* mOrd  = (unsigned*)alloc((size_t)N * HEADS * 4);
  float*    denom = (float*)   alloc((size_t)N * HEADS * 4);
  float*    deg   = (float*)   alloc((size_t)N * 4);
  unsigned* pmax  = (unsigned*)alloc((size_t)GRAPHS * F * 4);
  float*    psum  = (float*)   alloc((size_t)GRAPHS * F * 4);
  float*    pcnt  = (float*)   alloc((size_t)GRAPHS * 4);
  float*    G     = (float*)   alloc((size_t)GRAPHS * POOLF * 4);
  float*    G1    = (float*)   alloc((size_t)GRAPHS * FC1N * 4);
  half_t*   A16   = (half_t*)  alloc((size_t)N * KpG * 2);       // largest A
  half_t*   Wlp   = (half_t*)  alloc((size_t)Kp1 * Np1 * 2);
  half_t*   Wrp   = (half_t*)  alloc((size_t)Kp1 * Np1 * 2);
  half_t*   Wgp   = (half_t*)  alloc((size_t)KpG * NpG * 2);
  half_t*   Wf1p  = (half_t*)  alloc((size_t)Kf1 * Nf1 * 2);
  half_t*   Wf2p  = (half_t*)  alloc((size_t)Kf2 * Nf2 * 2);
  float*    Hg    = XR;   // XR dead after attn_logits
  float*    H2    = XL;   // XL dead after attn_agg

  // ---- weight packing (fragment-order f16) ----
  packB<<<blks((long long)Np1 * Kp1, 256), 256, 0, stream>>>(Wl,  Wlp,  CH,    F,    Kp1, Np1);
  packB<<<blks((long long)Np1 * Kp1, 256), 256, 0, stream>>>(Wr,  Wrp,  CH,    F,    Kp1, Np1);
  packB<<<blks((long long)NpG * KpG, 256), 256, 0, stream>>>(Wg,  Wgp,  F,     F,    KpG, NpG);
  packB<<<blks((long long)Nf1 * Kf1, 256), 256, 0, stream>>>(Wf1, Wf1p, POOLF, FC1N, Kf1, Nf1);
  packB<<<blks((long long)Nf2 * Kf2, 256), 256, 0, stream>>>(Wf2, Wf2p, FC1N,  FC2N, Kf2, Nf2);

  // ---- init ----
  fill_f32<<<blks((long long)N * HEADS, 256), 256, 0, stream>>>((float*)mOrd, 0.f, (long long)N * HEADS);
  fill_f32<<<blks((long long)N * HEADS, 256), 256, 0, stream>>>(denom, 0.f, (long long)N * HEADS);
  fill_f32<<<blks((long long)N * F, 256), 256, 0, stream>>>(H1, 0.f, (long long)N * F);
  fill_f32<<<blks(N, 256), 256, 0, stream>>>(deg, 0.f, N);

  // ---- GATv2: node transforms (WMMA GEMMs, shared f16 A) ----
  convA<<<blks((long long)N * Kp1, 256), 256, 0, stream>>>(x, A16, N, CH, Kp1);
  wmma_gemm<<<gemm_grid(N, Np1), 256, 0, stream>>>(A16, Wlp, nullptr, XL, N, F, Kp1, 0);
  wmma_gemm<<<gemm_grid(N, Np1), 256, 0, stream>>>(A16, Wrp, nullptr, XR, N, F, Kp1, 0);

  // ---- attention softmax over incoming edges ----
  attn_logits<<<blks(Etot, 8), 256, 0, stream>>>(XL, XR, ei, att, P, mOrd, E, N);
  attn_exp<<<blks((long long)Etot * HEADS, 256), 256, 0, stream>>>(ei, mOrd, P, denom, E, N);
  attn_agg<<<Etot, 256, 0, stream>>>(XL, ei, P, denom, H1, E, N);
  bias_relu<<<blks((long long)N * F, 256), 256, 0, stream>>>(H1, bgat, N, F, 1);

  // ---- GCN ----
  deg_kernel<<<blks(Etot, 256), 256, 0, stream>>>(ei, deg, E, N);
  convA<<<blks((long long)N * KpG, 256), 256, 0, stream>>>(H1, A16, N, F, KpG);
  wmma_gemm<<<gemm_grid(N, NpG), 256, 0, stream>>>(A16, Wgp, nullptr, Hg, N, F, KpG, 0);
  fill_f32<<<blks((long long)N * F, 256), 256, 0, stream>>>(H2, 0.f, (long long)N * F);
  gcn_agg<<<Etot, 256, 0, stream>>>(Hg, deg, ei, H2, E, N);
  bias_relu<<<blks((long long)N * F, 256), 256, 0, stream>>>(H2, bg, N, F, 1);

  // ---- global max/mean pool ----
  fill_f32<<<blks((long long)GRAPHS * F, 256), 256, 0, stream>>>((float*)pmax, 0.f, (long long)GRAPHS * F);
  fill_f32<<<blks((long long)GRAPHS * F, 256), 256, 0, stream>>>(psum, 0.f, (long long)GRAPHS * F);
  fill_f32<<<blks(GRAPHS, 256), 256, 0, stream>>>(pcnt, 0.f, GRAPHS);
  pool_cnt<<<blks(N, 256), 256, 0, stream>>>(batch, pcnt, N);
  pool_accum<<<blks((long long)N * F, 256), 256, 0, stream>>>(H2, batch, pmax, psum, N);
  pool_finalize<<<blks((long long)GRAPHS * F, 256), 256, 0, stream>>>(pmax, psum, pcnt, G);

  // ---- FC head (WMMA GEMMs) ----
  convA<<<blks((long long)GRAPHS * Kf1, 256), 256, 0, stream>>>(G, A16, GRAPHS, POOLF, Kf1);
  wmma_gemm<<<gemm_grid(GRAPHS, Nf1), 256, 0, stream>>>(A16, Wf1p, bf1, G1, GRAPHS, FC1N, Kf1, 1);
  convA<<<blks((long long)GRAPHS * Kf2, 256), 256, 0, stream>>>(G1, A16, GRAPHS, FC1N, Kf2);
  wmma_gemm<<<gemm_grid(GRAPHS, Nf2), 256, 0, stream>>>(A16, Wf2p, bf2, out, GRAPHS, FC2N, Kf2, 0);

  (void)n_in; (void)out_size; (void)ws_size;
}
